// MTCModel_38920993636390
// MI455X (gfx1250) — matrix operand order repaired and checked by reading, not compiled, not run on hardware
//
#include <hip/hip_runtime.h>

typedef __attribute__((ext_vector_type(16))) _Float16 v16h;
typedef __attribute__((ext_vector_type(8)))  _Float16 v8h;
typedef __attribute__((ext_vector_type(8)))  float    v8f;

#define BATCH   8192
#define NFIELD  26
#define M26     (BATCH * NFIELD)   // 212992 rows for the AutoInt GEMMs

#define BM 128
#define LDA 40   // padded LDS row (halves) for A tile: keeps 16B chunks aligned
#define LDB 40   // padded LDS row (halves) for transposed B tile

// ------------------------------------------------------------------
// f32 -> f16 weight conversion
// ------------------------------------------------------------------
__global__ void k_cvt(const float* __restrict__ s, _Float16* __restrict__ d, size_t n) {
  size_t i = (size_t)blockIdx.x * 256 + threadIdx.x;
  if (i < n) d[i] = (_Float16)s[i];
}

// ------------------------------------------------------------------
// Embedding gather (f16) + linear term
// ------------------------------------------------------------------
__global__ void k_embed(const int* __restrict__ x, const float* __restrict__ emb,
                        const float* __restrict__ lw, const float* __restrict__ lb,
                        _Float16* __restrict__ Xh, float* __restrict__ lin) {
  int b = blockIdx.x;
  __shared__ int idxs[NFIELD];
  __shared__ float lws[NFIELD];
  int t = threadIdx.x;
  if (t < NFIELD) {
    int id = x[b * NFIELD + t] + t * 1000;
    idxs[t] = id;
    lws[t] = lw[id];
  }
  __syncthreads();
  for (int e = t; e < NFIELD * 32; e += 256) {
    int f = e >> 5, d = e & 31;
    Xh[(size_t)b * 832 + e] = (_Float16)emb[(size_t)idxs[f] * 32 + d];
  }
  if (t == 0) {
    float s = lb[0];
#pragma unroll
    for (int f = 0; f < NFIELD; ++f) s += lws[f];
    lin[b] = s;
  }
}

// ------------------------------------------------------------------
// A tile (128 x 32 halves, row-major): per-lane 16B async copy to LDS.
// GLOBAL_LOAD_ASYNC_TO_LDS_B128: VDST = LDS byte offset (low 32 bits of the
// generic pointer to __shared__), VADDR = 64-bit global address; ASYNCcnt.
// All our M are multiples of 128, so no bounds checks needed.
// ------------------------------------------------------------------
__device__ __forceinline__ void load_A_async(const _Float16* __restrict__ X,
                                             _Float16* __restrict__ As,
                                             int m0, int k0, int K, int t) {
#pragma unroll
  for (int c = t; c < 512; c += 256) {
    int row = c >> 2;
    int col = (c & 3) << 3;
    const _Float16* gp = X + (size_t)(m0 + row) * K + k0 + col;
    unsigned lds = (unsigned)(size_t)(As + row * LDA + col);
    asm volatile("global_load_async_to_lds_b128 %0, %1, off"
                 :: "v"(lds), "v"(gp) : "memory");
  }
}

// B tile: 32 k-rows x BNT n-cols, staged transposed [n][k] so the per-lane
// B fragment is two contiguous ds_load_b128 (K-halves contiguous per column).
template <int BNT, bool FULLN>
__device__ __forceinline__ void load_B(const _Float16* __restrict__ We,
                                       _Float16* __restrict__ Bs,
                                       int n0, int k0, int N, int t) {
  constexpr int CPK = BNT / 8;   // 8-half chunks per k-row
#pragma unroll
  for (int c = t; c < 4 * BNT; c += 256) {
    int kk = c / CPK;
    int nn = (c % CPK) * 8;
    v8h val = {};
    if (FULLN || (n0 + nn < N))
      val = *(const v8h*)(We + (size_t)(k0 + kk) * N + n0 + nn);
#pragma unroll
    for (int q = 0; q < 8; ++q) Bs[(nn + q) * LDB + kk] = val[q];
  }
}

// ------------------------------------------------------------------
// Generic batched-expert WMMA GEMM:  Y[e] = act(X_e @ W[e] + bias[e])
//   X_e = Xa + e*xsE   (per-expert inputs, block-diagonal layers), or
//   X_e = group-select Xa/Xb/Xc by e/4 (shared / task0 / task1 inputs)
// NT = 16-wide n-tiles per wave (4 -> 128x128 block, 2 -> 128x64 block).
// ------------------------------------------------------------------
template <int NT, bool FULLN>
__global__ void __launch_bounds__(256)
k_gemm(const _Float16* __restrict__ Xa, const _Float16* __restrict__ Xb,
       const _Float16* __restrict__ Xc, long long xsE,
       const _Float16* __restrict__ W, const float* __restrict__ bias,
       _Float16* __restrict__ Y, int M, int N, int K, int relu) {
  constexpr int BNT = NT * 32;
  __shared__ __align__(16) _Float16 As[2][BM * LDA];
  __shared__ __align__(16) _Float16 Bs[2][BNT * LDB];
  int e = blockIdx.z;
  const _Float16* X = xsE ? (Xa + (long long)e * xsE)
                          : (e < 4 ? Xa : (e < 8 ? Xb : Xc));
  const _Float16* We = W + (size_t)e * K * N;
  int m0 = blockIdx.y * BM, n0 = blockIdx.x * BNT;
  int t = threadIdx.x, lane = t & 31, wid = t >> 5;
  int wm = wid >> 1, wn = wid & 1;          // 4x2 wave grid
  int lh = lane >> 4, ln = lane & 15;

  v8f acc[2][NT] = {};
  int nT = K >> 5;                          // all K here are multiples of 32
  load_A_async(X, As[0], m0, 0, K, t);
  load_B<BNT, FULLN>(We, Bs[0], n0, 0, N, t);
  asm volatile("s_wait_asynccnt 0x0" ::: "memory");
  __syncthreads();
  for (int kt = 0; kt < nT; ++kt) {
    int cur = kt & 1;
    if (kt + 1 < nT) {
      load_A_async(X, As[cur ^ 1], m0, (kt + 1) << 5, K, t);
      load_B<BNT, FULLN>(We, Bs[cur ^ 1], n0, (kt + 1) << 5, N, t);
    }
    const _Float16* as = &As[cur][0];
    const _Float16* bs = &Bs[cur][0];
    v16h af[2], bf[NT];
#pragma unroll
    for (int i = 0; i < 2; ++i) {
      // A frag: row m = wave row + lane%16; K halves (lane<16?0:8)+{0..7} and +16
      const _Float16* pa = as + (wm * 32 + i * 16 + ln) * LDA + (lh << 3);
      v8h lo = *(const v8h*)pa;
      v8h hi = *(const v8h*)(pa + 16);
      af[i] = __builtin_shufflevector(lo, hi, 0,1,2,3,4,5,6,7,8,9,10,11,12,13,14,15);
    }
#pragma unroll
    for (int j = 0; j < NT; ++j) {
      // B frag (column-major staged): col n = lane%16; K halves (lane<16?0:8)+{0..7} and +16
      const _Float16* pb = bs + (wn * (16 * NT) + j * 16 + ln) * LDB + (lh << 3);
      v8h lo = *(const v8h*)pb;
      v8h hi = *(const v8h*)(pb + 16);
      bf[j] = __builtin_shufflevector(lo, hi, 0,1,2,3,4,5,6,7,8,9,10,11,12,13,14,15);
    }
#pragma unroll
    for (int i = 0; i < 2; ++i)
#pragma unroll
      for (int j = 0; j < NT; ++j)
        acc[i][j] = __builtin_amdgcn_wmma_f32_16x16x32_f16(
            false, af[i], false, bf[j], (short)0, acc[i][j], false, false);
    asm volatile("s_wait_asynccnt 0x0" ::: "memory");
    __syncthreads();
  }
  // epilogue: C/D layout -> n = lane%16, m = vgpr + (lane>=16 ? 8 : 0)
  int rb = lh << 3;
#pragma unroll
  for (int j = 0; j < NT; ++j) {
    int gn = n0 + wn * (16 * NT) + j * 16 + ln;
    if (!FULLN && gn >= N) continue;
    float bv = bias ? bias[(size_t)e * N + gn] : 0.f;
#pragma unroll
    for (int i = 0; i < 2; ++i) {
#pragma unroll
      for (int r = 0; r < 8; ++r) {
        int gm = m0 + wm * 32 + i * 16 + rb + r;
        float v = acc[i][j][r] + bv;
        if (relu) v = fmaxf(v, 0.f);
        Y[(size_t)e * M * N + (size_t)gm * N + gn] = (_Float16)v;
      }
    }
  }
}

// ------------------------------------------------------------------
// AutoInt attention core: one wave per (batch, head); 26 tokens, hd=32
// ------------------------------------------------------------------
__global__ void k_attn(const _Float16* __restrict__ q, const _Float16* __restrict__ k,
                       const _Float16* __restrict__ v, _Float16* __restrict__ o) {
  int bh = blockIdx.x;
  int b = bh >> 1, h = bh & 1;
  size_t base = (size_t)b * NFIELD * 64 + h * 32;
  __shared__ float qs[NFIELD][32], ks[NFIELD][32], vs[NFIELD][32];
  int t = threadIdx.x;   // 0..31
  for (int i = 0; i < NFIELD; ++i) {
    qs[i][t] = (float)q[base + i * 64 + t];
    ks[i][t] = (float)k[base + i * 64 + t];
    vs[i][t] = (float)v[base + i * 64 + t];
  }
  __syncthreads();
  if (t < NFIELD) {
    float s[NFIELD];
    float mx = -1e30f;
    for (int j = 0; j < NFIELD; ++j) {
      float d = 0.f;
#pragma unroll
      for (int dd = 0; dd < 32; ++dd) d += qs[t][dd] * ks[j][dd];
      d *= 0.17677669529663687f;   // 1/sqrt(32)
      s[j] = d; mx = fmaxf(mx, d);
    }
    float den = 0.f;
    for (int j = 0; j < NFIELD; ++j) { s[j] = __expf(s[j] - mx); den += s[j]; }
    float inv = 1.f / den;
#pragma unroll
    for (int dd = 0; dd < 32; ++dd) {
      float a = 0.f;
      for (int j = 0; j < NFIELD; ++j) a += s[j] * vs[j][dd];
      o[base + t * 64 + dd] = (_Float16)(a * inv);
    }
  }
}

// ------------------------------------------------------------------
// cross = relu(a_final + vres) then atten_out = cross @ fw + fb
// ------------------------------------------------------------------
__global__ void k_fc(const _Float16* __restrict__ af, const _Float16* __restrict__ vf,
                     const float* __restrict__ fw, const float* __restrict__ fb,
                     float* __restrict__ out) {
  int b = blockIdx.x;
  __shared__ float red[256];
  float s = 0.f;
  for (int i = threadIdx.x; i < 1664; i += 256) {
    float c = (float)af[(size_t)b * 1664 + i] + (float)vf[(size_t)b * 1664 + i];
    c = fmaxf(c, 0.f);
    s += c * fw[i];
  }
  red[threadIdx.x] = s;
  __syncthreads();
  for (int o = 128; o > 0; o >>= 1) {
    if (threadIdx.x < o) red[threadIdx.x] += red[threadIdx.x + o];
    __syncthreads();
  }
  if (threadIdx.x == 0) out[b] = red[0] + fb[0];
}

// ------------------------------------------------------------------
// gate = softmax(X @ gw + gb) — one wave32 per row, shfl reduction
// ------------------------------------------------------------------
__global__ void k_gate(const _Float16* __restrict__ X, const float* __restrict__ gw,
                       const float* __restrict__ gb, float* __restrict__ out,
                       int Kd, int G) {
  int b = blockIdx.x * 8 + (threadIdx.x >> 5);
  int lane = threadIdx.x & 31;
  float acc[12];
#pragma unroll
  for (int g = 0; g < 12; ++g) acc[g] = 0.f;
  const _Float16* xr = X + (size_t)b * Kd;
  for (int k = lane; k < Kd; k += 32) {
    float xv = (float)xr[k];
    for (int g = 0; g < G; ++g) acc[g] += xv * gw[(size_t)k * G + g];
  }
  for (int g = 0; g < G; ++g)
    for (int o = 16; o > 0; o >>= 1) acc[g] += __shfl_down(acc[g], o, 32);
  if (lane == 0) {
    float mx = -1e30f;
    for (int g = 0; g < G; ++g) { acc[g] += gb[g]; mx = fmaxf(mx, acc[g]); }
    float den = 0.f;
    for (int g = 0; g < G; ++g) { acc[g] = __expf(acc[g] - mx); den += acc[g]; }
    float inv = 1.f / den;
    for (int g = 0; g < G; ++g) out[(size_t)b * G + g] = acc[g] * inv;
  }
}

// ------------------------------------------------------------------
// gated mix: Y[b,d] = sum_g gate[b,g] * O_group(g)[(g&3), b, d]
//   g 0..3 -> Oa, 4..7 -> Ob, 8..11 -> Oc
// ------------------------------------------------------------------
__global__ void k_mix(const _Float16* __restrict__ Oa, const _Float16* __restrict__ Ob,
                      const _Float16* __restrict__ Oc, const float* __restrict__ gates,
                      int G, int nshift, long long strideE,
                      _Float16* __restrict__ Y, int total) {
  int i = blockIdx.x * 256 + threadIdx.x;
  if (i >= total) return;
  int b = i >> nshift;
  const float* g = gates + (size_t)b * G;
  float s = 0.f;
  for (int e = 0; e < G; ++e) {
    const _Float16* O = (e < 4) ? Oa : ((e < 8) ? Ob : Oc);
    s += g[e] * (float)O[(long long)(e & 3) * strideE + i];
  }
  Y[i] = (_Float16)s;
}

// ------------------------------------------------------------------
// Output heads: r0 (sigmoid), r1, final = r1 - united
// ------------------------------------------------------------------
__global__ void k_heads(const float* __restrict__ lin, const float* __restrict__ att,
                        const _Float16* __restrict__ t0, const _Float16* __restrict__ t1,
                        const _Float16* __restrict__ tu,
                        const float* __restrict__ w0a, const float* __restrict__ b0a,
                        const float* __restrict__ w0b, const float* __restrict__ b0b,
                        const float* __restrict__ w1a, const float* __restrict__ b1a,
                        const float* __restrict__ w1b, const float* __restrict__ b1b,
                        const float* __restrict__ wua, const float* __restrict__ bua,
                        const float* __restrict__ wub, const float* __restrict__ bub,
                        float* __restrict__ out, int B) {
  int b = blockIdx.x * 256 + threadIdx.x;
  if (b >= B) return;
  float f[34];
  f[0] = lin[b]; f[1] = att[b];
  for (int i = 0; i < 32; ++i) f[2 + i] = (float)t0[(size_t)b * 32 + i];
  float h8[8];
  for (int o = 0; o < 8; ++o) {
    float s = b0a[o];
    for (int i = 0; i < 34; ++i) s += f[i] * w0a[i * 8 + o];
    h8[o] = s;
  }
  float z = b0b[0];
  for (int o = 0; o < 8; ++o) z += h8[o] * w0b[o];
  out[b] = 1.f / (1.f + __expf(-z));

  for (int i = 0; i < 32; ++i) f[2 + i] = (float)t1[(size_t)b * 32 + i];
  float h9[9], r1[9];
  for (int o = 0; o < 9; ++o) {
    float s = b1a[o];
    for (int i = 0; i < 34; ++i) s += f[i] * w1a[i * 9 + o];
    h9[o] = s;
  }
  for (int o = 0; o < 9; ++o) {
    float s = b1b[o];
    for (int i = 0; i < 9; ++i) s += h9[i] * w1b[i * 9 + o];
    r1[o] = s;
    out[B + (size_t)b * 9 + o] = s;
  }
  float u9[9];
  for (int o = 0; o < 9; ++o) {
    float s = bua[o];
    for (int i = 0; i < 32; ++i) s += (float)tu[(size_t)b * 32 + i] * wua[i * 9 + o];
    u9[o] = s;
  }
  for (int o = 0; o < 9; ++o) {
    float s = bub[o];
    for (int i = 0; i < 9; ++i) s += u9[i] * wub[i * 9 + o];
    out[(size_t)10 * B + (size_t)b * 9 + o] = r1[o] - s;
  }
}

// ==================================================================
extern "C" void kernel_launch(void* const* d_in, const int* in_sizes, int n_in,
                              void* d_out, int out_size, void* d_ws, size_t ws_size,
                              hipStream_t stream) {
  (void)in_sizes; (void)n_in; (void)out_size;
  char* ws = (char*)d_ws;
  const size_t B = BATCH;
  size_t off = 0;
  auto alloc = [&](size_t bytes) { size_t r = off; off = (off + bytes + 255) & ~(size_t)255; return r; };

  // persistent buffers + packed f16 weights
  size_t o_Xh    = alloc(B * 832 * 2);
  size_t o_lin   = alloc(B * 4);
  size_t o_atn   = alloc(B * 4);
  size_t o_attWh = alloc(2048 * 2);
  size_t o_qkvo  = alloc((size_t)12 * 4096 * 2);
  size_t o_vresWh= alloc(2048 * 2);
  size_t o_W0L0  = alloc((size_t)12 * 832 * 512 * 2);
  size_t o_W1L0  = alloc((size_t)12 * 512 * 256 * 2);
  size_t o_W0L1  = alloc((size_t)12 * 256 * 256 * 2);
  size_t o_W1L1  = alloc((size_t)12 * 256 * 128 * 2);
  size_t o_tw0a  = alloc(8192 * 2), o_tw0b = alloc(2048 * 2);
  size_t o_tw1a  = alloc(8192 * 2), o_tw1b = alloc(2048 * 2);
  size_t o_twua  = alloc(8192 * 2), o_twub = alloc(2048 * 2);
  size_t o_b0L0  = alloc(12 * 512 * 4);
  size_t o_b1L0  = alloc(12 * 256 * 4);
  size_t o_b0L1  = alloc(12 * 256 * 4);
  size_t o_b1L1  = alloc(12 * 128 * 4);

  // attention scratch (phase A) and PLE scratch (phase B) share one region
  const size_t SZA = (size_t)M26 * 64 * 2;  // one (B*26,64) f16 buffer
  size_t attn_total = 7 * SZA;
  size_t p = 0;
  auto palloc = [&](size_t bytes) { size_t r = p; p = (p + bytes + 255) & ~(size_t)255; return r; };
  size_t p_H0  = palloc((size_t)12 * B * 512 * 2);
  size_t p_O0  = palloc((size_t)12 * B * 256 * 2);
  size_t p_g0t0 = palloc(B * 8 * 4), p_g0t1 = palloc(B * 8 * 4), p_g0s = palloc(B * 12 * 4);
  size_t p_X10 = palloc(B * 256 * 2), p_X11 = palloc(B * 256 * 2), p_X1s = palloc(B * 256 * 2);
  size_t p_H1  = palloc((size_t)12 * B * 256 * 2);
  size_t p_O1  = palloc((size_t)12 * B * 128 * 2);
  size_t p_g1t0 = palloc(B * 8 * 4), p_g1t1 = palloc(B * 8 * 4), p_g1s = palloc(B * 4 * 4);
  size_t p_F10 = palloc(B * 128 * 2), p_F11 = palloc(B * 128 * 2), p_F1s = palloc(B * 128 * 2);
  size_t p_T0a = palloc(B * 64 * 2),  p_T0 = palloc(B * 32 * 2);
  size_t p_T1a = palloc(B * 64 * 2),  p_T1 = palloc(B * 32 * 2);
  size_t p_TUa = palloc(B * 64 * 2),  p_TU = palloc(B * 32 * 2);
  size_t region = alloc(attn_total > p ? attn_total : p);
  if (off > ws_size) return;

  auto H = [&](size_t o) { return (_Float16*)(ws + o); };
  auto F = [&](size_t o) { return (float*)(ws + o); };
  auto IN = [&](int i) { return (const float*)d_in[i]; };

  _Float16* a_h  = H(region + 0 * SZA);
  _Float16* q_h  = H(region + 1 * SZA);
  _Float16* k_h  = H(region + 2 * SZA);
  _Float16* v_h  = H(region + 3 * SZA);
  _Float16* o_h  = H(region + 4 * SZA);
  _Float16* af_h = H(region + 5 * SZA);
  _Float16* vf_h = H(region + 6 * SZA);

  auto cvt = [&](int src, size_t dst, size_t n) {
    k_cvt<<<dim3((unsigned)((n + 255) / 256)), dim3(256), 0, stream>>>(IN(src), H(dst), n);
  };
  auto cpy = [&](size_t dst, int src, size_t nf) {
    hipMemcpyAsync(ws + dst, d_in[src], nf * 4, hipMemcpyDeviceToDevice, stream);
  };
  auto gemm = [&](const _Float16* Xa, const _Float16* Xb, const _Float16* Xc, long long xsE,
                  const _Float16* W, const float* bias, _Float16* Y,
                  int M, int N, int K, int relu, int E) {
    if (N >= 128) {          // N multiple of 128: full 128x128 tiles
      dim3 g((unsigned)(N / 128), (unsigned)(M / 128), (unsigned)E);
      k_gemm<4, true><<<g, dim3(256), 0, stream>>>(Xa, Xb, Xc, xsE, W, bias, Y, M, N, K, relu);
    } else if (N == 64) {    // full 128x64 tiles
      dim3 g(1, (unsigned)(M / 128), (unsigned)E);
      k_gemm<2, true><<<g, dim3(256), 0, stream>>>(Xa, Xb, Xc, xsE, W, bias, Y, M, N, K, relu);
    } else {                 // N=32 towers: guarded 128x64 tile
      dim3 g(1, (unsigned)(M / 128), (unsigned)E);
      k_gemm<2, false><<<g, dim3(256), 0, stream>>>(Xa, Xb, Xc, xsE, W, bias, Y, M, N, K, relu);
    }
  };

  // ---- weight conversion / packing (expert order: shared 0-3, task0 4-7, task1 8-11)
  cvt(4, o_attWh, 2048);
  cvt(30, o_vresWh, 2048);
  for (int l = 0; l < 3; ++l)
    for (int m = 0; m < 4; ++m)
      cvt(6 + 8 * l + 2 * m, o_qkvo + (size_t)(l * 4 + m) * 4096 * 2, 4096);
  const size_t s00 = (size_t)4 * 832 * 512, s01 = (size_t)4 * 512 * 256;
  const size_t s10 = (size_t)4 * 256 * 256, s11 = (size_t)4 * 256 * 128;
  cvt(34, o_W0L0, s00); cvt(42, o_W0L0 + s00 * 2, s00); cvt(46, o_W0L0 + 2 * s00 * 2, s00);
  cvt(36, o_W1L0, s01); cvt(44, o_W1L0 + s01 * 2, s01); cvt(48, o_W1L0 + 2 * s01 * 2, s01);
  cvt(38, o_W0L1, s10); cvt(50, o_W0L1 + s10 * 2, s10); cvt(54, o_W0L1 + 2 * s10 * 2, s10);
  cvt(40, o_W1L1, s11); cvt(52, o_W1L1 + s11 * 2, s11); cvt(56, o_W1L1 + 2 * s11 * 2, s11);
  cvt(70, o_tw0a, 8192); cvt(72, o_tw0b, 2048);
  cvt(74, o_tw1a, 8192); cvt(76, o_tw1b, 2048);
  cvt(78, o_twua, 8192); cvt(80, o_twub, 2048);
  cpy(o_b0L0, 35, 2048); cpy(o_b0L0 + 2048 * 4, 43, 2048); cpy(o_b0L0 + 4096 * 4, 47, 2048);
  cpy(o_b1L0, 37, 1024); cpy(o_b1L0 + 1024 * 4, 45, 1024); cpy(o_b1L0 + 2048 * 4, 49, 1024);
  cpy(o_b0L1, 39, 1024); cpy(o_b0L1 + 1024 * 4, 51, 1024); cpy(o_b0L1 + 2048 * 4, 55, 1024);
  cpy(o_b1L1, 41, 512);  cpy(o_b1L1 + 512 * 4, 53, 512);   cpy(o_b1L1 + 1024 * 4, 57, 512);

  // ---- embedding + linear term
  k_embed<<<dim3(BATCH), dim3(256), 0, stream>>>(
      (const int*)d_in[0], IN(1), IN(2), IN(3), H(o_Xh), F(o_lin));

  // ---- AutoInt: a = embed @ attW + b, then 3 attention layers
  gemm(H(o_Xh), H(o_Xh), H(o_Xh), 0, H(o_attWh), IN(5), a_h, M26, 64, 32, 0, 1);
  for (int l = 0; l < 3; ++l) {
    const _Float16* wq = H(o_qkvo) + (size_t)(l * 4 + 0) * 4096;
    const _Float16* wk = H(o_qkvo) + (size_t)(l * 4 + 1) * 4096;
    const _Float16* wv = H(o_qkvo) + (size_t)(l * 4 + 2) * 4096;
    const _Float16* wo = H(o_qkvo) + (size_t)(l * 4 + 3) * 4096;
    gemm(a_h, a_h, a_h, 0, wq, IN(7 + 8 * l),  q_h, M26, 64, 64, 0, 1);
    gemm(a_h, a_h, a_h, 0, wk, IN(9 + 8 * l),  k_h, M26, 64, 64, 0, 1);
    gemm(a_h, a_h, a_h, 0, wv, IN(11 + 8 * l), v_h, M26, 64, 64, 0, 1);
    k_attn<<<dim3(BATCH * 2), dim3(32), 0, stream>>>(q_h, k_h, v_h, o_h);
    gemm(o_h, o_h, o_h, 0, wo, IN(13 + 8 * l), (l == 2) ? af_h : a_h, M26, 64, 64, 0, 1);
  }
  gemm(H(o_Xh), H(o_Xh), H(o_Xh), 0, H(o_vresWh), IN(31), vf_h, M26, 64, 32, 0, 1);
  k_fc<<<dim3(BATCH), dim3(256), 0, stream>>>(af_h, vf_h, IN(32), IN(33), F(o_atn));

  // ---- PLE level 0 (all gates see ei): 12 experts 832->512->256
  _Float16* H0 = H(region + p_H0); _Float16* O0 = H(region + p_O0);
  k_gate<<<dim3(BATCH / 8), dim3(256), 0, stream>>>(H(o_Xh), IN(58), IN(59), F(region + p_g0t0), 832, 8);
  k_gate<<<dim3(BATCH / 8), dim3(256), 0, stream>>>(H(o_Xh), IN(60), IN(61), F(region + p_g0t1), 832, 8);
  k_gate<<<dim3(BATCH / 8), dim3(256), 0, stream>>>(H(o_Xh), IN(66), IN(67), F(region + p_g0s), 832, 12);
  gemm(H(o_Xh), H(o_Xh), H(o_Xh), 0, H(o_W0L0), F(o_b0L0), H0, BATCH, 512, 832, 1, 12);
  gemm(H0, H0, H0, (long long)B * 512, H(o_W1L0), F(o_b1L0), O0, BATCH, 256, 512, 1, 12);
  {
    long long sE = (long long)B * 256; int tot = (int)(B * 256);
    dim3 g((unsigned)(tot / 256));
    k_mix<<<g, 256, 0, stream>>>(O0 + 4 * sE, O0, O0, F(region + p_g0t0), 8, 8, sE, H(region + p_X10), tot);
    k_mix<<<g, 256, 0, stream>>>(O0 + 8 * sE, O0, O0, F(region + p_g0t1), 8, 8, sE, H(region + p_X11), tot);
    k_mix<<<g, 256, 0, stream>>>(O0 + 4 * sE, O0 + 8 * sE, O0, F(region + p_g0s), 12, 8, sE, H(region + p_X1s), tot);
  }

  // ---- PLE level 1: 12 experts 256->256->128 (inputs differ per group)
  _Float16* X10 = H(region + p_X10); _Float16* X11 = H(region + p_X11); _Float16* X1s = H(region + p_X1s);
  _Float16* H1 = H(region + p_H1); _Float16* O1 = H(region + p_O1);
  k_gate<<<dim3(BATCH / 8), dim3(256), 0, stream>>>(X10, IN(62), IN(63), F(region + p_g1t0), 256, 8);
  k_gate<<<dim3(BATCH / 8), dim3(256), 0, stream>>>(X11, IN(64), IN(65), F(region + p_g1t1), 256, 8);
  k_gate<<<dim3(BATCH / 8), dim3(256), 0, stream>>>(X1s, IN(68), IN(69), F(region + p_g1s), 256, 4);
  gemm(X1s, X10, X11, 0, H(o_W0L1), F(o_b0L1), H1, BATCH, 256, 256, 1, 12);
  gemm(H1, H1, H1, (long long)B * 256, H(o_W1L1), F(o_b1L1), O1, BATCH, 128, 256, 1, 12);
  {
    long long sE = (long long)B * 128; int tot = (int)(B * 128);
    dim3 g((unsigned)(tot / 256));
    k_mix<<<g, 256, 0, stream>>>(O1 + 4 * sE, O1, O1, F(region + p_g1t0), 8, 7, sE, H(region + p_F10), tot);
    k_mix<<<g, 256, 0, stream>>>(O1 + 8 * sE, O1, O1, F(region + p_g1t1), 8, 7, sE, H(region + p_F11), tot);
    k_mix<<<g, 256, 0, stream>>>(O1, O1, O1, F(region + p_g1s), 4, 7, sE, H(region + p_F1s), tot);
  }

  // ---- towers (WMMA): 128->64->32
  _Float16* F10 = H(region + p_F10); _Float16* F11 = H(region + p_F11); _Float16* F1s = H(region + p_F1s);
  _Float16* T0a = H(region + p_T0a); _Float16* T0 = H(region + p_T0);
  _Float16* T1a = H(region + p_T1a); _Float16* T1 = H(region + p_T1);
  _Float16* TUa = H(region + p_TUa); _Float16* TU = H(region + p_TU);
  gemm(F10, F10, F10, 0, H(o_tw0a), IN(71), T0a, BATCH, 64, 128, 1, 1);
  gemm(T0a, T0a, T0a, 0, H(o_tw0b), IN(73), T0,  BATCH, 32, 64, 1, 1);
  gemm(F11, F11, F11, 0, H(o_tw1a), IN(75), T1a, BATCH, 64, 128, 1, 1);
  gemm(T1a, T1a, T1a, 0, H(o_tw1b), IN(77), T1,  BATCH, 32, 64, 1, 1);
  gemm(F1s, F1s, F1s, 0, H(o_twua), IN(79), TUa, BATCH, 64, 128, 1, 1);
  gemm(TUa, TUa, TUa, 0, H(o_twub), IN(81), TU,  BATCH, 32, 64, 1, 1);

  // ---- output heads
  k_heads<<<dim3(BATCH / 256), dim3(256), 0, stream>>>(
      F(o_lin), F(o_atn), T0, T1, TU,
      IN(82), IN(83), IN(84), IN(85),
      IN(86), IN(87), IN(88), IN(89),
      IN(90), IN(91), IN(92), IN(93),
      (float*)d_out, BATCH);
}
